// VQTokenizer_53120155517394
// MI455X (gfx1250) — matrix operand order repaired and checked by reading, not compiled, not run on hardware
//
#include <hip/hip_runtime.h>
#include <stdint.h>

typedef __attribute__((ext_vector_type(16))) _Float16 v16h;
typedef __attribute__((ext_vector_type(8)))  float    v8f;

union Frag {
    uint32_t u[8];
    v16h     h;
};

// A-fragment (16x32 f16) K-pair base for VGPR v, lane-half 'half'
__device__ __forceinline__ int kkA(int v, int half) {
    return ((v & 3) << 1) + (half << 3) + ((v >> 2) << 4);
}
// B-fragment (32x16 f16) K-pair base for VGPR v, lane-half 'half'
__device__ __forceinline__ int kkB(int v, int half) {
    return ((v & 3) << 1) + (half << 3) + ((v >> 2) << 4);
}

#define WMMA_F16(A, B, C) \
    __builtin_amdgcn_wmma_f32_16x16x32_f16(false, (A), false, (B), (short)0, (C), false, false)

// ---------------------------------------------------------------------------
// prep kernels
// ---------------------------------------------------------------------------
__global__ __launch_bounds__(256) void cvt_f16_kernel(const float* __restrict__ in,
                                                      _Float16* __restrict__ out, int n) {
    int i = blockIdx.x * 256 + threadIdx.x;
    if (i < n) out[i] = (_Float16)in[i];
}

__global__ __launch_bounds__(256) void cb_halfnorm_kernel(const float* __restrict__ cb,
                                                          float* __restrict__ hn) {
    int k = blockIdx.x * 256 + threadIdx.x;   // 8192 rows
    const float* row = cb + (size_t)k * 256;
    float s = 0.f;
    #pragma unroll 8
    for (int j = 0; j < 256; ++j) { float v = row[j]; s += v * v; }
    hn[k] = 0.5f * s;
}

// dec_w [256][768] f32  ->  dec_wT [768][256] f16
__global__ __launch_bounds__(256) void cvt_decwT_kernel(const float* __restrict__ in,
                                                        _Float16* __restrict__ out) {
    int o = blockIdx.x * 256 + threadIdx.x;   // 196608
    int d = o & 255;
    int j = o >> 8;
    out[o] = (_Float16)in[d * 768 + j];
}

// ---------------------------------------------------------------------------
// Stage 1: encoder  z[n,d] = patch(n) . enc_w[d] + enc_b[d]   (stored f16)
// One wave per 16x16 tile of z[16384 x 256]; K = 768 (24 wmma steps).
// ---------------------------------------------------------------------------
__global__ __launch_bounds__(256) void enc_kernel(const float* __restrict__ x,
                                                  const _Float16* __restrict__ w16,
                                                  const float* __restrict__ bvec,
                                                  _Float16* __restrict__ z16) {
    const int wave = (blockIdx.x << 3) + (threadIdx.x >> 5);   // 0..16383
    const int lane = threadIdx.x & 31;
    const int half = lane >> 4, l16 = lane & 15;
    const int mBase = (wave >> 4) << 4;
    const int nBase = (wave & 15) << 4;

    const int row = mBase + l16;                 // patch/token index
    const int b   = row >> 10;
    const int rem = row & 1023;
    const int hl16 = ((rem >> 5) << 4);
    const int wl16 = ((rem & 31) << 4);
    const int b3   = b * 3;

    const int dcol = nBase + l16;                // output feature (column)
    const float bias = bvec[dcol];
    const _Float16* wrow = w16 + (size_t)dcol * 768;

    v8f acc = {};
    for (int kt = 0; kt < 24; ++kt) {
        const int kBase = kt << 5;
        Frag A, B;
        #pragma unroll
        for (int v = 0; v < 8; ++v) {
            const int k = kBase + kkA(v, half);
            const int c = k >> 8, r2 = k & 255, p = r2 >> 4, q = r2 & 15;
            const float2 f =
                *(const float2*)(x + ((((size_t)(b3 + c) << 9) + hl16 + p) << 9) + wl16 + q);
            A.h[2 * v]     = (_Float16)f.x;
            A.h[2 * v + 1] = (_Float16)f.y;
            B.u[v] = *(const uint32_t*)(wrow + kBase + kkB(v, half));
        }
        acc = WMMA_F16(A.h, B.h, acc);
    }
    #pragma unroll
    for (int r = 0; r < 8; ++r) {
        const int m = mBase + (half << 3) + r;
        z16[(size_t)m * 256 + dcol] = (_Float16)(acc[r] + bias);
    }
}

// ---------------------------------------------------------------------------
// Stage 2: fused distance GEMM + partial argmax.
// score[n,k] = z_n . c_k - 0.5*||c_k||^2 ; argmax_k == argmin squared dist.
// Each wave: 32 rows (two 16-row tiles => two independent WMMA chains for
// co-execution) x a 1/4 slice of the 8192 columns (2048 waves total).
// Partial (max, argmax) written per (16-row tile, sub-slice); dec_kernel
// finishes the 4-way reduction.
// ---------------------------------------------------------------------------
__global__ __launch_bounds__(256) void vq_kernel(const _Float16* __restrict__ z16,
                                                 const _Float16* __restrict__ cb16,
                                                 const float* __restrict__ hn,
                                                 float* __restrict__ pval,
                                                 int* __restrict__ pidx) {
    const int wave = (blockIdx.x << 3) + (threadIdx.x >> 5);   // 0..2047
    const int lane = threadIdx.x & 31;
    const int half = lane >> 4, l16 = lane & 15;
    const int mSuper = wave >> 2;                              // 0..511
    const int sub    = wave & 3;                               // column slice
    const int mBase  = mSuper << 5;                            // 32 rows

    Frag a[2][8];
    #pragma unroll
    for (int rt = 0; rt < 2; ++rt) {
        const _Float16* zrow = z16 + (size_t)(mBase + (rt << 4) + l16) * 256;
        #pragma unroll
        for (int kt = 0; kt < 8; ++kt)
            #pragma unroll
            for (int v = 0; v < 8; ++v)
                a[rt][kt].u[v] = *(const uint32_t*)(zrow + (kt << 5) + kkA(v, half));
    }

    float best[2][8];
    int   bidx[2][8];
    #pragma unroll
    for (int rt = 0; rt < 2; ++rt)
        #pragma unroll
        for (int r = 0; r < 8; ++r) { best[rt][r] = -3.4e38f; bidx[rt][r] = 0; }

    const int ctBeg = sub << 7;                 // 128 column tiles per slice
    const int ctEnd = ctBeg + 128;
    for (int ct = ctBeg; ct < ctEnd; ++ct) {
        const int col = (ct << 4) + l16;
        __builtin_prefetch(cb16 + (size_t)((col + 16) & 8191) * 256, 0, 0);
        const float cbias = hn[col];
        const _Float16* crow = cb16 + (size_t)col * 256;

        v8f acc0 = {}, acc1 = {};
        #pragma unroll
        for (int kt = 0; kt < 8; ++kt) {
            Frag B;
            #pragma unroll
            for (int v = 0; v < 8; ++v)
                B.u[v] = *(const uint32_t*)(crow + (kt << 5) + kkB(v, half));
            acc0 = WMMA_F16(a[0][kt].h, B.h, acc0);   // independent chains:
            acc1 = WMMA_F16(a[1][kt].h, B.h, acc1);   // B reused, 2 row tiles
        }
        #pragma unroll
        for (int r = 0; r < 8; ++r) {
            const float s0 = acc0[r] - cbias;
            const float s1 = acc1[r] - cbias;
            if (s0 > best[0][r]) { best[0][r] = s0; bidx[0][r] = col; }
            if (s1 > best[1][r]) { best[1][r] = s1; bidx[1][r] = col; }
        }
    }

    #pragma unroll
    for (int rt = 0; rt < 2; ++rt)
        #pragma unroll
        for (int r = 0; r < 8; ++r) {
            float bv = best[rt][r];
            int   bi = bidx[rt][r];
            #pragma unroll
            for (int off = 8; off >= 1; off >>= 1) {
                const float ov = __shfl_xor(bv, off, 32);
                const int   oi = __shfl_xor(bi, off, 32);
                if (ov > bv || (ov == bv && oi < bi)) { bv = ov; bi = oi; }
            }
            if (l16 == 0) {
                const int t    = (mBase >> 4) + rt;               // 16-row tile id
                const int slot = (t << 6) + (sub << 4) + (half << 3) + r;
                pval[slot] = bv;
                pidx[slot] = bi;
            }
        }
}

// ---------------------------------------------------------------------------
// Stage 3: decoder. Resolves the 4-way argmax partials per row, then
// y = codebook[idx] @ dec_w + dec_b scattered to NCHW.
// Each wave: one 16-row tile x 12 of 48 feature tiles, processed in pairs
// with dual accumulators (4096 waves).
// ---------------------------------------------------------------------------
__global__ __launch_bounds__(256) void dec_kernel(const float* __restrict__ pval,
                                                  const int* __restrict__ pidx,
                                                  const _Float16* __restrict__ cb16,
                                                  const _Float16* __restrict__ wT16,
                                                  const float* __restrict__ bvec,
                                                  float* __restrict__ y) {
    const int wave = (blockIdx.x << 3) + (threadIdx.x >> 5);   // 0..4095
    const int lane = threadIdx.x & 31;
    const int half = lane >> 4, l16 = lane & 15;
    const int mTile = wave >> 2;                               // 0..1023
    const int sub   = wave & 3;                                // feature slice
    const int mBase = mTile << 4;

    // finish the argmax reduction for this lane's row (ascending column
    // slices + strict '>' == lowest index on ties, matching argmin-first)
    const int pbase = (mTile << 6) + l16;
    float bv   = pval[pbase];
    int   code = pidx[pbase];
    #pragma unroll
    for (int s = 1; s < 4; ++s) {
        const float v = pval[pbase + (s << 4)];
        const int   i = pidx[pbase + (s << 4)];
        if (v > bv) { bv = v; code = i; }
    }

    Frag a[8];
    const _Float16* crow = cb16 + (size_t)code * 256;
    #pragma unroll
    for (int kt = 0; kt < 8; ++kt)
        #pragma unroll
        for (int v = 0; v < 8; ++v)
            a[kt].u[v] = *(const uint32_t*)(crow + (kt << 5) + kkA(v, half));

    for (int jj = 0; jj < 6; ++jj) {
        const int jB0 = ((sub * 12) + (jj << 1)) << 4;
        const int jB1 = jB0 + 16;
        const int cc0 = jB0 >> 8, pp0 = (jB0 >> 4) & 15;
        const int cc1 = jB1 >> 8, pp1 = (jB1 >> 4) & 15;
        const float bias0 = bvec[cc0];
        const float bias1 = bvec[cc1];
        const _Float16* w0 = wT16 + (size_t)(jB0 + l16) * 256;
        const _Float16* w1 = wT16 + (size_t)(jB1 + l16) * 256;

        v8f acc0 = {}, acc1 = {};
        #pragma unroll
        for (int kt = 0; kt < 8; ++kt) {
            Frag B0, B1;
            #pragma unroll
            for (int v = 0; v < 8; ++v) {
                B0.u[v] = *(const uint32_t*)(w0 + (kt << 5) + kkB(v, half));
                B1.u[v] = *(const uint32_t*)(w1 + (kt << 5) + kkB(v, half));
            }
            acc0 = WMMA_F16(a[kt].h, B0.h, acc0);
            acc1 = WMMA_F16(a[kt].h, B1.h, acc1);
        }
        #pragma unroll
        for (int r = 0; r < 8; ++r) {
            const int n   = mBase + (half << 3) + r;
            const int b   = n >> 10;
            const int rem = n & 1023;
            const int hl  = rem >> 5, wl = rem & 31;
            const size_t base = (((size_t)(b * 3) << 9) << 9)
                              + ((size_t)(hl << 4) << 9) + (wl << 4) + l16;
            y[base + (((size_t)cc0 << 9) << 9) + ((size_t)pp0 << 9)] = acc0[r] + bias0;
            y[base + (((size_t)cc1 << 9) << 9) + ((size_t)pp1 << 9)] = acc1[r] + bias1;
        }
    }
}

// ---------------------------------------------------------------------------
extern "C" void kernel_launch(void* const* d_in, const int* in_sizes, int n_in,
                              void* d_out, int out_size, void* d_ws, size_t ws_size,
                              hipStream_t stream) {
    const float* x    = (const float*)d_in[0];
    const float* encw = (const float*)d_in[1];
    const float* encb = (const float*)d_in[2];
    const float* cb   = (const float*)d_in[3];
    const float* decw = (const float*)d_in[4];
    const float* decb = (const float*)d_in[5];
    float* y = (float*)d_out;

    // workspace carve-up (~13.9 MB)
    char* ws = (char*)d_ws;
    _Float16* encw16 = (_Float16*)(ws + 0x000000);            // 256*768*2   = 0x060000
    _Float16* cb16   = (_Float16*)(ws + 0x060000);            // 8192*256*2  = 0x400000
    float*    hn     = (float*)   (ws + 0x460000);            // 8192*4      = 0x008000
    _Float16* wT16   = (_Float16*)(ws + 0x468000);            // 768*256*2   = 0x060000
    _Float16* z16    = (_Float16*)(ws + 0x4C8000);            // 16384*256*2 = 0x800000
    float*    pval   = (float*)   (ws + 0xCC8000);            // 1024*64*4   = 0x040000
    int*      pidx   = (int*)     (ws + 0xD08000);            // 1024*64*4   = 0x040000

    cvt_f16_kernel    <<<768,  256, 0, stream>>>(encw, encw16, 256 * 768);
    cvt_f16_kernel    <<<8192, 256, 0, stream>>>(cb,   cb16,   8192 * 256);
    cb_halfnorm_kernel<<<32,   256, 0, stream>>>(cb, hn);
    cvt_decwT_kernel  <<<768,  256, 0, stream>>>(decw, wT16);

    enc_kernel<<<2048, 256, 0, stream>>>(x, encw16, encb, z16);       // 16384 waves
    vq_kernel <<<256,  256, 0, stream>>>(z16, cb16, hn, pval, pidx);  // 2048 waves
    dec_kernel<<<512,  256, 0, stream>>>(pval, pidx, cb16, wT16, decb, y); // 4096 waves
}
